// NTMCell_60833916781026
// MI455X (gfx1250) — compile-verified, matrix operand order checked by reading, not run on hardware
//
#include <hip/hip_runtime.h>
#include <hip/hip_bf16.h>
#include <math.h>

// ---------------- problem constants ----------------
#define BDIM   64
#define IDIM   10
#define MDIM   64
#define ADIM   16384
#define HCD    256
#define HRD    2
#define ODIM   8
#define PDIM   338        // HR*(M+6) + 3M+6
#define KZ     394        // I + HR*M + HC
#define KZP    416        // KZ padded to mult of 32
#define PPAD   352        // PDIM padded to mult of 16
#define GDIM   1024       // 4*HC
#define EPSV   1e-8f

typedef __attribute__((ext_vector_type(16))) _Float16 v16h;
typedef __attribute__((ext_vector_type(8)))  float    v8f;

__device__ __forceinline__ float sigmoidf_(float x) { return 1.0f / (1.0f + expf(-x)); }
__device__ __forceinline__ float softplusf_(float x) { return (x > 20.0f) ? x : log1pf(expf(x)); }

// ---------------- packing kernels (f32 -> f16 staging for WMMA) ----------------
__global__ void ntm_pack_z(const float* __restrict__ inputs,
                           const float* __restrict__ read_vecs,
                           const float* __restrict__ h,
                           _Float16* __restrict__ z16) {
    int idx = blockIdx.x * blockDim.x + threadIdx.x;
    if (idx >= BDIM * KZP) return;
    int b = idx / KZP, k = idx % KZP;
    float v = 0.0f;
    if (k < IDIM)                 v = inputs[b * IDIM + k];
    else if (k < IDIM + HRD*MDIM) v = read_vecs[b * (HRD*MDIM) + (k - IDIM)];
    else if (k < KZ)              v = h[b * HCD + (k - IDIM - HRD*MDIM)];
    z16[idx] = (_Float16)v;
}

__global__ void ntm_pack_wcat(const float* __restrict__ W_ih,
                              const float* __restrict__ W_hh,
                              _Float16* __restrict__ w16) {
    int idx = blockIdx.x * blockDim.x + threadIdx.x;
    if (idx >= GDIM * KZP) return;
    int n = idx / KZP, k = idx % KZP;
    float v = 0.0f;
    if (k < IDIM + HRD*MDIM) v = W_ih[n * (IDIM + HRD*MDIM) + k];
    else if (k < KZ)         v = W_hh[n * HCD + (k - IDIM - HRD*MDIM)];
    w16[idx] = (_Float16)v;
}

__global__ void ntm_pack_wint(const float* __restrict__ W_int, _Float16* __restrict__ w16) {
    int idx = blockIdx.x * blockDim.x + threadIdx.x;
    if (idx >= PPAD * HCD) return;
    int n = idx / HCD, k = idx % HCD;
    w16[idx] = (_Float16)((n < PDIM) ? W_int[n * HCD + k] : 0.0f);
}

// ---------------- WMMA GEMM ----------------
// C = A(MxK,f16,row-major,lda) * W^T (W is [N x K], f16, row-major, ldw) + bias
// Tiles are scattered into LDS in *fragment order*: lane l's 16 halves are
// contiguous (FA[l*16..l*16+15]) so each lane loads its fragment with 2x
// ds_load_b128. Mapping per ISA 7.12.2 (16-bit A 16x32 / B 32x16):
//   lane = row + 16*((k>>3)&1), slot = (k&7) + 8*((k>>4)&1)
template<int KSTEPS>
__global__ void ntm_wmma_gemm(const _Float16* __restrict__ Amat, int lda,
                              const _Float16* __restrict__ Wmat, int ldw,
                              const float* __restrict__ bias,
                              float* __restrict__ Cmat, int ldc, int Nvalid) {
    __shared__ __align__(32) _Float16 FA[32 * 16];
    __shared__ __align__(32) _Float16 FB[32 * 16];
    const int lane = threadIdx.x;     // blockDim.x == 32 (one wave)
    const int m0 = blockIdx.x * 16;
    const int n0 = blockIdx.y * 16;
    const int g  = lane >> 4;
    const int ln = lane & 15;
    v8f acc = {};
    #pragma unroll
    for (int ks = 0; ks < KSTEPS; ++ks) {
        const int k0 = ks * 32;
        for (int idx = lane; idx < 512; idx += 32) {
            int m = idx >> 5, k = idx & 31;          // A[m][k]
            int dl = m + ((k & 8) << 1);
            int sl = (k & 7) + ((k >> 4) << 3);
            FA[dl * 16 + sl] = Amat[(m0 + m) * lda + (k0 + k)];
        }
        for (int idx = lane; idx < 512; idx += 32) {
            int k = idx >> 4, n = idx & 15;          // B[k][n] = W[n][k]
            int dl = n + ((k & 8) << 1);
            int sl = (k & 7) + ((k >> 4) << 3);
            FB[dl * 16 + sl] = Wmat[(n0 + n) * ldw + (k0 + k)];
        }
        __syncthreads();
        v16h a = *(const v16h*)&FA[lane * 16];
        v16h b = *(const v16h*)&FB[lane * 16];
        acc = __builtin_amdgcn_wmma_f32_16x16x32_f16(
                  false, a, false, b, (short)0, acc, false, false);
        __syncthreads();
    }
    if (n0 + ln < Nvalid) {
        float bv = bias[n0 + ln];
        #pragma unroll
        for (int v = 0; v < 8; ++v)
            Cmat[(m0 + v + 8 * g) * ldc + (n0 + ln)] = acc[v] + bv;
    }
}

// ---------------- LSTM pointwise ----------------
__global__ void ntm_lstm(const float* __restrict__ gates, const float* __restrict__ c_in,
                         float* __restrict__ h_out, float* __restrict__ c_out,
                         _Float16* __restrict__ h16) {
    int idx = blockIdx.x * blockDim.x + threadIdx.x;
    if (idx >= BDIM * HCD) return;
    int b = idx / HCD, j = idx % HCD;
    const float* gr = gates + b * GDIM;
    float ig = gr[j], fg = gr[HCD + j], gg = gr[2 * HCD + j], og = gr[3 * HCD + j];
    float cn = sigmoidf_(fg) * c_in[idx] + sigmoidf_(ig) * tanhf(gg);
    float hn = sigmoidf_(og) * tanhf(cn);
    c_out[idx] = cn;
    h_out[idx] = hn;
    h16[idx] = (_Float16)hn;
}

// ---------------- logits ----------------
__global__ void ntm_logits(const float* __restrict__ h_new, const float* __restrict__ W_out,
                           const float* __restrict__ b_out, float* __restrict__ out) {
    int idx = blockIdx.x * blockDim.x + threadIdx.x;
    if (idx >= BDIM * ODIM) return;
    int b = idx / ODIM, o = idx % ODIM;
    float acc = b_out[o];
    for (int j = 0; j < HCD; ++j) acc += h_new[b * HCD + j] * W_out[o * HCD + j];
    out[idx] = acc;
}

// ---------------- normalized keys + erase/add precompute ----------------
__global__ void ntm_keys(const float* __restrict__ p, float* __restrict__ kn_w,
                         float* __restrict__ kn_r, float* __restrict__ er,
                         float* __restrict__ ad) {
    const int b = blockIdx.x;
    const int t = threadIdx.x;  // 64 threads
    __shared__ float red[64];
    const float* pb = p + b * PPAD;
    for (int r = 0; r < HRD; ++r) {
        float v = pb[r * (MDIM + 6) + t];
        red[t] = v * v; __syncthreads();
        for (int s = 32; s > 0; s >>= 1) { if (t < s) red[t] += red[t + s]; __syncthreads(); }
        float inv = 1.0f / (sqrtf(red[0]) + EPSV); __syncthreads();
        kn_r[(r * BDIM + b) * MDIM + t] = v * inv;
    }
    float v = pb[HRD * (MDIM + 6) + t];
    red[t] = v * v; __syncthreads();
    for (int s = 32; s > 0; s >>= 1) { if (t < s) red[t] += red[t + s]; __syncthreads(); }
    float inv = 1.0f / (sqrtf(red[0]) + EPSV); __syncthreads();
    kn_w[b * MDIM + t] = v * inv;
    er[b * MDIM + t] = sigmoidf_(pb[HRD * (MDIM + 6) + MDIM + 6 + t]);
    ad[b * MDIM + t] = pb[HRD * (MDIM + 6) + 2 * MDIM + 6 + t];
}

// ---------------- write-head content sim (float4 stream of original memory) ----------------
__global__ void ntm_simw(const float* __restrict__ mem, const float* __restrict__ kn_w,
                         float* __restrict__ sim_w) {
    const int b  = blockIdx.y;
    const int a4 = blockIdx.x * 256 + threadIdx.x;   // float4 index; gridDim.x = ADIM/1024
    __shared__ float kk[MDIM];
    if (threadIdx.x < MDIM) kk[threadIdx.x] = kn_w[b * MDIM + threadIdx.x];
    __syncthreads();
    const float4* mb = (const float4*)(mem + (size_t)b * MDIM * ADIM);
    const int stride4 = ADIM / 4;
    float dot[4] = {0, 0, 0, 0}, ss[4] = {0, 0, 0, 0};
    #pragma unroll 4
    for (int m = 0; m < MDIM; ++m) {
        __builtin_prefetch(mb + (size_t)(m + 4) * stride4 + a4, 0, 3);
        float4 v = mb[(size_t)m * stride4 + a4];
        const float* vp = (const float*)&v;
        #pragma unroll
        for (int j = 0; j < 4; ++j) { dot[j] += kk[m] * vp[j]; ss[j] += vp[j] * vp[j]; }
    }
    float4 o;
    float* op = (float*)&o;
    #pragma unroll
    for (int j = 0; j < 4; ++j) op[j] = dot[j] / (sqrtf(ss[j]) + EPSV);
    ((float4*)(sim_w + (size_t)b * ADIM))[a4] = o;
}

// ---------------- generic addressing: softmax -> interp -> shift -> sharpen ----------------
__global__ void ntm_address(const float* __restrict__ p, int poff,
                            const float* __restrict__ sim, int sim_stride_b,
                            const float* __restrict__ w_prev, int wprev_stride_b,
                            float* __restrict__ w_out, int wout_stride_b) {
    const int b = blockIdx.x;
    const int t = threadIdx.x;          // 256 threads
    const int CH = ADIM / 256;          // 64 per thread
    __shared__ float red[256];
    __shared__ float bndL[256], bndR[256];
    const float* pb = p + b * PPAD;
    float beta = softplusf_(pb[poff + MDIM]);
    float gg   = sigmoidf_(pb[poff + MDIM + 1]);
    float s0r = pb[poff + MDIM + 2], s1r = pb[poff + MDIM + 3], s2r = pb[poff + MDIM + 4];
    float smx = fmaxf(s0r, fmaxf(s1r, s2r));
    float e0 = expf(s0r - smx), e1 = expf(s1r - smx), e2 = expf(s2r - smx);
    float einv = 1.0f / (e0 + e1 + e2);
    float s0 = e0 * einv, s1 = e1 * einv, s2 = e2 * einv;
    float gamma = 1.0f + softplusf_(pb[poff + MDIM + 5]);

    const float* simb = sim + (size_t)b * sim_stride_b;
    const float* wpb  = w_prev + (size_t)b * wprev_stride_b;
    float v[64];
    const int a0 = t * CH;

    float lmax = -INFINITY;
    #pragma unroll
    for (int j = 0; j < 64; ++j) { v[j] = beta * simb[a0 + j]; lmax = fmaxf(lmax, v[j]); }
    red[t] = lmax; __syncthreads();
    for (int s = 128; s > 0; s >>= 1) { if (t < s) red[t] = fmaxf(red[t], red[t + s]); __syncthreads(); }
    float gmax = red[0]; __syncthreads();

    float lsum = 0.0f;
    #pragma unroll
    for (int j = 0; j < 64; ++j) { v[j] = expf(v[j] - gmax); lsum += v[j]; }
    red[t] = lsum; __syncthreads();
    for (int s = 128; s > 0; s >>= 1) { if (t < s) red[t] += red[t + s]; __syncthreads(); }
    float inv = 1.0f / red[0]; __syncthreads();

    #pragma unroll
    for (int j = 0; j < 64; ++j) {
        float wc = v[j] * inv;
        v[j] = gg * wc + (1.0f - gg) * wpb[a0 + j];
    }
    bndL[t] = v[0]; bndR[t] = v[63]; __syncthreads();
    float prev = bndR[(t + 255) & 255];   // wg[a0-1] (circular)
    float next = bndL[(t + 1) & 255];     // wg[a0+CH] (circular)

    lsum = 0.0f;
    float wgm1 = prev;
    #pragma unroll
    for (int j = 0; j < 64; ++j) {
        float cur  = v[j];
        float wgp1 = (j == 63) ? next : v[j + 1];
        float ws   = s0 * wgp1 + s1 * cur + s2 * wgm1;
        float wp   = powf(ws + EPSV, gamma);
        v[j] = wp; lsum += wp; wgm1 = cur;
    }
    red[t] = lsum; __syncthreads();
    for (int s = 128; s > 0; s >>= 1) { if (t < s) red[t] += red[t + s]; __syncthreads(); }
    float invs = 1.0f / (red[0] + EPSV);
    float* wo = w_out + (size_t)b * wout_stride_b;
    #pragma unroll
    for (int j = 0; j < 64; ++j) wo[a0 + j] = v[j] * invs;
}

// ---------------- memory update fused with read-head sims (float4 stream) ----------------
__global__ void ntm_memupd(const float* __restrict__ mem, const float* __restrict__ w_wr,
                           const float* __restrict__ kn_r, const float* __restrict__ er,
                           const float* __restrict__ ad, float* __restrict__ mem_new,
                           float* __restrict__ sim_r) {
    const int b  = blockIdx.y;
    const int a4 = blockIdx.x * 256 + threadIdx.x;   // float4 index
    const int t  = threadIdx.x;
    __shared__ float k0[MDIM], k1[MDIM], se[MDIM], sa[MDIM];
    if (t < 64)       k0[t]       = kn_r[(0 * BDIM + b) * MDIM + t];
    else if (t < 128) k1[t - 64]  = kn_r[(1 * BDIM + b) * MDIM + (t - 64)];
    else if (t < 192) se[t - 128] = er[b * MDIM + (t - 128)];
    else              sa[t - 192] = ad[b * MDIM + (t - 192)];
    __syncthreads();
    const float4 w4 = ((const float4*)(w_wr + (size_t)b * ADIM))[a4];
    const float* wp = (const float*)&w4;
    const float4* mb = (const float4*)(mem + (size_t)b * MDIM * ADIM);
    float4* ob = (float4*)(mem_new + (size_t)b * MDIM * ADIM);
    const int stride4 = ADIM / 4;
    float d0[4] = {0,0,0,0}, d1[4] = {0,0,0,0}, ss[4] = {0,0,0,0};
    #pragma unroll 4
    for (int m = 0; m < MDIM; ++m) {
        __builtin_prefetch(mb + (size_t)(m + 4) * stride4 + a4, 0, 3);
        float4 v = mb[(size_t)m * stride4 + a4];
        const float* vv = (const float*)&v;
        float4 nvv;
        float* np = (float*)&nvv;
        #pragma unroll
        for (int j = 0; j < 4; ++j) {
            float nv = vv[j] * (1.0f - se[m] * wp[j]) + sa[m] * wp[j];
            np[j] = nv;
            ss[j] += nv * nv; d0[j] += k0[m] * nv; d1[j] += k1[m] * nv;
        }
        ob[(size_t)m * stride4 + a4] = nvv;
    }
    float4 s0v, s1v;
    float* s0p = (float*)&s0v; float* s1p = (float*)&s1v;
    #pragma unroll
    for (int j = 0; j < 4; ++j) {
        float invn = 1.0f / (sqrtf(ss[j]) + EPSV);
        s0p[j] = d0[j] * invn; s1p[j] = d1[j] * invn;
    }
    ((float4*)(sim_r + (size_t)(0 * BDIM + b) * ADIM))[a4] = s0v;
    ((float4*)(sim_r + (size_t)(1 * BDIM + b) * ADIM))[a4] = s1v;
}

// ---------------- read vectors via async global->LDS pipeline ----------------
// rv[b][r][m] = sum_a mem_new[b][m][a] * w[b][r][a]
__global__ void ntm_readvec(const float* __restrict__ mem_new, const float* __restrict__ rw,
                            float* __restrict__ rv) {
    const int gid = blockIdx.x;                 // B*HR*M blocks
    const int m = gid % MDIM;
    const int r = (gid / MDIM) % HRD;
    const int b = gid / (MDIM * HRD);
    const float* row = mem_new + ((size_t)b * MDIM + m) * ADIM;
    const float* w   = rw + ((size_t)b * HRD + r) * ADIM;
    __shared__ __align__(16) float stage[2][2048];
    __shared__ float red[256];
    const int t = threadIdx.x;
    constexpr int NCH = ADIM / 2048;            // 8 chunks

    auto issue = [&](int buf, int chunk) {
        // 2048 floats per chunk: 256 threads x 2 x b128 async-to-LDS
        const float* src0 = row + chunk * 2048 + t * 4;
        unsigned lds0 = (unsigned)(uintptr_t)&stage[buf][t * 4];
        asm volatile("global_load_async_to_lds_b128 %0, %1, off"
                     :: "v"(lds0), "v"(src0) : "memory");
        const float* src1 = src0 + 1024;
        unsigned lds1 = (unsigned)(uintptr_t)&stage[buf][1024 + t * 4];
        asm volatile("global_load_async_to_lds_b128 %0, %1, off"
                     :: "v"(lds1), "v"(src1) : "memory");
    };

    float acc = 0.0f;
    issue(0, 0);
    for (int chunk = 0; chunk < NCH; ++chunk) {
        if (chunk + 1 < NCH) {
            issue((chunk + 1) & 1, chunk + 1);
            asm volatile("s_wait_asynccnt 0x2" ::: "memory");  // current chunk landed
        } else {
            asm volatile("s_wait_asynccnt 0x0" ::: "memory");
        }
        __syncthreads();
        const float* sb = stage[chunk & 1];
        const float* wb = w + chunk * 2048;
        #pragma unroll 4
        for (int i = t; i < 2048; i += 256) acc += sb[i] * wb[i];
        __syncthreads();
    }
    red[t] = acc; __syncthreads();
    for (int s = 128; s > 0; s >>= 1) { if (t < s) red[t] += red[t + s]; __syncthreads(); }
    if (t == 0) rv[((size_t)b * HRD + r) * MDIM + m] = red[0];
}

// ---------------- host launcher ----------------
extern "C" void kernel_launch(void* const* d_in, const int* in_sizes, int n_in,
                              void* d_out, int out_size, void* d_ws, size_t ws_size,
                              hipStream_t stream) {
    (void)in_sizes; (void)n_in; (void)out_size; (void)ws_size;
    const float* inputs    = (const float*)d_in[0];
    const float* memory    = (const float*)d_in[1];
    const float* read_w    = (const float*)d_in[2];
    const float* write_w   = (const float*)d_in[3];
    const float* h_in      = (const float*)d_in[4];
    const float* c_in      = (const float*)d_in[5];
    const float* read_vecs = (const float*)d_in[6];
    const float* W_ih      = (const float*)d_in[7];
    const float* W_hh      = (const float*)d_in[8];
    const float* b_lstm    = (const float*)d_in[9];
    const float* W_int     = (const float*)d_in[10];
    const float* b_int     = (const float*)d_in[11];
    const float* W_out     = (const float*)d_in[12];
    const float* b_out     = (const float*)d_in[13];

    float* out = (float*)d_out;
    float* o_logits = out;                                   // 512
    float* o_mem    = out + 512;                             // 67,108,864
    float* o_rv     = out + 512 + 67108864ull;               // 8,192
    float* o_rw     = out + 512 + 67108864ull + 8192;        // 2,097,152
    float* o_wwr    = o_rw + 2097152;                        // 1,048,576
    float* o_h      = o_wwr + 1048576;                       // 16,384
    float* o_c      = o_h + 16384;                           // 16,384

    char* wptr = (char*)d_ws;
    auto alloc = [&](size_t bytes) {
        char* rr = wptr;
        wptr += (bytes + 255) & ~(size_t)255;
        return rr;
    };
    _Float16* z16    = (_Float16*)alloc((size_t)BDIM * KZP * 2);
    _Float16* wcat16 = (_Float16*)alloc((size_t)GDIM * KZP * 2);
    _Float16* wint16 = (_Float16*)alloc((size_t)PPAD * HCD * 2);
    _Float16* h16    = (_Float16*)alloc((size_t)BDIM * HCD * 2);
    float* gates = (float*)alloc((size_t)BDIM * GDIM * 4);
    float* pbuf  = (float*)alloc((size_t)BDIM * PPAD * 4);
    float* kn_w  = (float*)alloc((size_t)BDIM * MDIM * 4);
    float* kn_r  = (float*)alloc((size_t)HRD * BDIM * MDIM * 4);
    float* erz   = (float*)alloc((size_t)BDIM * MDIM * 4);
    float* adz   = (float*)alloc((size_t)BDIM * MDIM * 4);
    float* sim_w = (float*)alloc((size_t)BDIM * ADIM * 4);
    float* sim_r = (float*)alloc((size_t)HRD * BDIM * ADIM * 4);

    // 1) pack f16 operands
    ntm_pack_z<<<(BDIM * KZP + 255) / 256, 256, 0, stream>>>(inputs, read_vecs, h_in, z16);
    ntm_pack_wcat<<<(GDIM * KZP + 255) / 256, 256, 0, stream>>>(W_ih, W_hh, wcat16);
    ntm_pack_wint<<<(PPAD * HCD + 255) / 256, 256, 0, stream>>>(W_int, wint16);

    // 2) gates = [x,h] @ [W_ih,W_hh]^T + b_lstm   (WMMA, K=416 fully unrolled)
    ntm_wmma_gemm<KZP / 32><<<dim3(BDIM / 16, GDIM / 16), 32, 0, stream>>>(
        z16, KZP, wcat16, KZP, b_lstm, gates, GDIM, GDIM);

    // 3) LSTM pointwise -> h_new, c_new (outputs) + f16 copy of h_new
    ntm_lstm<<<(BDIM * HCD + 255) / 256, 256, 0, stream>>>(gates, c_in, o_h, o_c, h16);

    // 4) p = h_new @ W_int^T + b_int  (WMMA, K=256 fully unrolled)
    ntm_wmma_gemm<HCD / 32><<<dim3(BDIM / 16, PPAD / 16), 32, 0, stream>>>(
        h16, HCD, wint16, HCD, b_int, pbuf, PPAD, PDIM);

    // 5) logits (tiny)
    ntm_logits<<<(BDIM * ODIM + 255) / 256, 256, 0, stream>>>(o_h, W_out, b_out, o_logits);

    // 6) normalized keys, erase, add
    ntm_keys<<<BDIM, 64, 0, stream>>>(pbuf, kn_w, kn_r, erz, adz);

    // 7) write-head content sim over original memory (256 MiB b128 stream)
    ntm_simw<<<dim3(ADIM / 1024, BDIM), 256, 0, stream>>>(memory, kn_w, sim_w);

    // 8) write-head addressing -> w_wr (output)
    ntm_address<<<BDIM, 256, 0, stream>>>(
        pbuf, HRD * (MDIM + 6), sim_w, ADIM, write_w, ADIM, o_wwr, ADIM);

    // 9) memory update + fused read-head sims (read 256 MiB, write 256 MiB, b128)
    ntm_memupd<<<dim3(ADIM / 1024, BDIM), 256, 0, stream>>>(
        memory, o_wwr, kn_r, erz, adz, o_mem, sim_r);

    // 10) read-head addressing -> read weights (output)
    for (int r = 0; r < HRD; ++r) {
        ntm_address<<<BDIM, 256, 0, stream>>>(
            pbuf, r * (MDIM + 6),
            sim_r + (size_t)r * BDIM * ADIM, ADIM,
            read_w + (size_t)r * ADIM, HRD * ADIM,
            o_rw + (size_t)r * ADIM, HRD * ADIM);
    }

    // 11) read vectors (512 MiB async-to-LDS pipelined stream)
    ntm_readvec<<<BDIM * HRD * MDIM, 256, 0, stream>>>(o_mem, o_rw, o_rv);
}
